// Layer1_34789235097686
// MI455X (gfx1250) — matrix operand order repaired and checked by reading, not compiled, not run on hardware
//
#include <hip/hip_runtime.h>

typedef _Float16 v16h __attribute__((ext_vector_type(16)));
typedef _Float16 h8   __attribute__((ext_vector_type(8)));
typedef float    v8f  __attribute__((ext_vector_type(8)));
typedef float    f4   __attribute__((ext_vector_type(4)));
typedef int      v4i_ __attribute__((vector_size(16)));

#define N_    8192
#define D_    128
#define DOUT_ 128
#define MB    128   // output rows per block
#define KC    64    // K per staged chunk (2 WMMA K-steps)

#if __has_builtin(__builtin_amdgcn_global_load_async_to_lds_b128) && \
    __has_builtin(__builtin_amdgcn_s_wait_asynccnt)
#define USE_ASYNC_LDS 1
#else
#define USE_ASYNC_LDS 0
#endif

__device__ __forceinline__ h8 cvt8(f4 a, f4 b) {
    h8 h;
    h[0] = (_Float16)a[0]; h[1] = (_Float16)a[1];
    h[2] = (_Float16)a[2]; h[3] = (_Float16)a[3];
    h[4] = (_Float16)b[0]; h[5] = (_Float16)b[1];
    h[6] = (_Float16)b[2]; h[7] = (_Float16)b[3];
    return h;
}

#if USE_ASYNC_LDS
__device__ __forceinline__ void async_copy_b128(const void* g, void* l) {
    __builtin_amdgcn_global_load_async_to_lds_b128(
        (__attribute__((address_space(1))) v4i_*)(uintptr_t)g,
        (__attribute__((address_space(3))) v4i_*)(unsigned)(uintptr_t)l,
        0, 0);
}
#endif

// ---------------------------------------------------------------------------
// Kernel 1: swizzle h0 (N x D fp32, row-major) into WMMA B-fragment layout
// (f16) in workspace: ws[t][j][lane][16], t = k/32 (256), j = col/16 (8).
// B-frag element (k, c): lane = (c%16) + 16*((k%32)>>4), f16 idx = k%16.
// ---------------------------------------------------------------------------
__global__ void h0_swizzle(const float* __restrict__ h0,
                           _Float16* __restrict__ ws) {
    unsigned gid = blockIdx.x * blockDim.x + threadIdx.x;   // 0 .. 524287
    int d    = gid & 7;
    int lane = (gid >> 3) & 31;
    int j    = (gid >> 8) & 7;
    int t    = gid >> 11;
    int k = t * 32 + ((lane >> 4) << 4) + 2 * d;
    int c = j * 16 + (lane & 15);
    _Float16 a = (_Float16)h0[(size_t)k * D_ + c];
    _Float16 b = (_Float16)h0[(size_t)(k + 1) * D_ + c];
    ws[(size_t)gid * 2]     = a;
    ws[(size_t)gid * 2 + 1] = b;
}

// ---------------------------------------------------------------------------
// Kernel 2: fused (rownorm(adj) @ h0) -> concat(self) -> relu(@ W^T) -> out
// grid = (64 M-blocks, 2 branches), 256 threads = 8 waves.
// A tiles: register double-buffered (global loads fly during WMMA phase).
// B tiles: async global->LDS double-buffer if available, else register path.
// ---------------------------------------------------------------------------
__global__ __launch_bounds__(256, 1) void gnn_fused(
    const float* __restrict__ adj_pos,
    const float* __restrict__ adj_neg,
    const float* __restrict__ h0,
    const float* __restrict__ WB0,
    const float* __restrict__ WU0,
    const _Float16* __restrict__ hB,   // pre-swizzled h0 fragments
    float* __restrict__ out)           // (N, DOUT, 1, 2)
{
    __shared__ _Float16 lds_w[8][8][32][16];                 // 64 KB W frags
    __shared__ __align__(16) unsigned char pool[67584];      // A(16K)+B(2x16K) ∪ feat
    __shared__ float rowsum[MB];

    _Float16 (*lds_a)[8][32][16] = (_Float16 (*)[8][32][16])pool;  // [2][8][32][16]
    _Float16 (*feat)[264]        = (_Float16 (*)[264])pool;        // [128][264] padded
    unsigned char* ldsb_base     = pool + 16384;                   // 2 x 16 KB

    const int tid  = threadIdx.x;
    const int lane = tid & 31;
    const int w    = tid >> 5;
    const int br   = blockIdx.y;
    const int row0 = blockIdx.x * MB;
    const float* __restrict__ adj = br ? adj_neg : adj_pos;
    const float* __restrict__ W   = br ? WU0 : WB0;

    const int r_loc = tid >> 1;     // 0..127: row within block this thread stages
    const int half  = tid & 1;      // which 32-wide K half of the chunk

    // ---- stage W into B-fragment layout (once): Bt[f][c] = W[c][f] ----
    {
        const int c = r_loc;        // 0..127 output column
        #pragma unroll
        for (int ss = 0; ss < 16; ++ss) {
            int f0 = half * 128 + ss * 8;
            f4 x0 = *(const f4*)(W + (size_t)c * 256 + f0);
            f4 x1 = *(const f4*)(W + (size_t)c * 256 + f0 + 4);
            h8 h = cvt8(x0, x1);
            int t      = f0 >> 5;
            int lane_w = (c & 15) + (((f0 >> 4) & 1) << 4);
            *(h8*)&lds_w[t][c >> 4][lane_w][f0 & 15] = h;
        }
        if (tid < MB) rowsum[tid] = 0.f;
    }

    v8f acc[8] = {};

    const float* srcA = adj + (size_t)(row0 + r_loc) * N_ + half * 32;
    const unsigned char* srcBb = (const unsigned char*)hB;   // chunk c at byte c*16384

    // ---- prologue: prefetch chunk 0 ----
    f4 pa[8];
    #pragma unroll
    for (int ss = 0; ss < 4; ++ss) {
        pa[2 * ss]     = *(const f4*)(srcA + ss * 8);
        pa[2 * ss + 1] = *(const f4*)(srcA + ss * 8 + 4);
    }
#if USE_ASYNC_LDS
    #pragma unroll
    for (int i = 0; i < 4; ++i)
        async_copy_b128(srcBb + (tid + i * 256) * 16,
                        ldsb_base + (tid + i * 256) * 16);
#else
    uint4 pb[4];
    #pragma unroll
    for (int i = 0; i < 4; ++i) pb[i] = ((const uint4*)srcBb)[tid + i * 256];
#endif

    __syncthreads();   // covers lds_w / rowsum init

    for (int kc0 = 0; kc0 < N_; kc0 += KC) {
        const int buf      = (kc0 >> 6) & 1;
        const bool has_nxt = (kc0 + KC) < N_;

        // ---- write staged A registers to LDS fragments, fused row-sum ----
        {
            float psum = 0.f;
            #pragma unroll
            for (int ss = 0; ss < 4; ++ss) {
                f4 x0 = pa[2 * ss], x1 = pa[2 * ss + 1];
                psum += x0[0] + x0[1] + x0[2] + x0[3] + x1[0] + x1[1] + x1[2] + x1[3];
                h8 h = cvt8(x0, x1);
                // A-frag (r,k): lane = r + 16*((k>>4)&1), idx = k%16
                *(h8*)&lds_a[half][r_loc >> 4][(r_loc & 15) + ((ss >> 1) << 4)][(ss & 1) * 8] = h;
            }
            atomicAdd(&rowsum[r_loc], psum);
        }
#if !USE_ASYNC_LDS
        {
            uint4* dst = (uint4*)(ldsb_base + buf * 16384);
            #pragma unroll
            for (int i = 0; i < 4; ++i) dst[tid + i * 256] = pb[i];
        }
#endif
        // ---- prefetch next chunk: loads overlap the whole WMMA phase ----
        if (has_nxt) {
            const float* sa = srcA + kc0 + KC;
            #pragma unroll
            for (int ss = 0; ss < 4; ++ss) {
                pa[2 * ss]     = *(const f4*)(sa + ss * 8);
                pa[2 * ss + 1] = *(const f4*)(sa + ss * 8 + 4);
            }
#if USE_ASYNC_LDS
            const unsigned char* sb = srcBb + (size_t)((kc0 + KC) >> 6) * 16384;
            unsigned char* db = ldsb_base + (buf ^ 1) * 16384;
            #pragma unroll
            for (int i = 0; i < 4; ++i)
                async_copy_b128(sb + (tid + i * 256) * 16,
                                db + (tid + i * 256) * 16);
#else
            const uint4* sb = (const uint4*)(srcBb + (size_t)((kc0 + KC) >> 6) * 16384);
            #pragma unroll
            for (int i = 0; i < 4; ++i) pb[i] = sb[tid + i * 256];
#endif
        }
#if USE_ASYNC_LDS
        // current chunk's 4 in-order async copies must have landed; keep the
        // next chunk's 4 in flight across the WMMA phase.
        if (has_nxt) __builtin_amdgcn_s_wait_asynccnt(4);
        else         __builtin_amdgcn_s_wait_asynccnt(0);
#endif
        __syncthreads();

        // ---- compute: 2 K-steps x 8 N-tiles, explicit B pipelining ----
        _Float16 (*ldsb)[32][16] = (_Float16 (*)[32][16])(ldsb_base + buf * 16384); // [2*8][32][16]
        v16h a0 = *(const v16h*)&lds_a[0][w][lane][0];
        v16h a1 = *(const v16h*)&lds_a[1][w][lane][0];
        #pragma unroll
        for (int s = 0; s < 2; ++s) {
            v16h a = s ? a1 : a0;
            v16h b0 = *(const v16h*)&ldsb[s * 8][lane][0];
            #pragma unroll
            for (int j = 0; j < 8; ++j) {
                v16h b1;
                if (j < 7) b1 = *(const v16h*)&ldsb[s * 8 + j + 1][lane][0];
                acc[j] = __builtin_amdgcn_wmma_f32_16x16x32_f16(
                    false, a, false, b0, (short)0, acc[j], false, false);
                if (j < 7) b0 = b1;
            }
        }
        __syncthreads();
    }

    // ---- tail: normalize, build feat = [agg | self] in LDS (f16) ----
    #pragma unroll
    for (int j = 0; j < 8; ++j) {
        #pragma unroll
        for (int v = 0; v < 8; ++v) {
            int r = w * 16 + v + ((lane >> 4) << 3);
            int c = j * 16 + (lane & 15);
            feat[r][c] = (_Float16)(acc[j][v] / rowsum[r]);
        }
    }
    {
        const float* src = h0 + (size_t)(row0 + r_loc) * D_ + half * 64;
        #pragma unroll
        for (int ss = 0; ss < 8; ++ss) {
            f4 x0 = *(const f4*)(src + ss * 8);
            f4 x1 = *(const f4*)(src + ss * 8 + 4);
            h8 h = cvt8(x0, x1);
            *(h8*)&feat[r_loc][128 + half * 64 + ss * 8] = h;
        }
    }
    __syncthreads();

    // ---- tail GEMM: out = relu(feat @ W^T), K = 256 ----
    v8f acc2[8] = {};
    #pragma unroll
    for (int t = 0; t < 8; ++t) {
        int r  = w * 16 + (lane & 15);
        int kb = t * 32 + ((lane >> 4) << 4);
        h8 lo = *(const h8*)&feat[r][kb];
        h8 hi = *(const h8*)&feat[r][kb + 8];
        v16h a;
        #pragma unroll
        for (int i = 0; i < 8; ++i) { a[i] = lo[i]; a[8 + i] = hi[i]; }
        #pragma unroll
        for (int j = 0; j < 8; ++j) {
            v16h b = *(const v16h*)&lds_w[t][j][lane][0];
            acc2[j] = __builtin_amdgcn_wmma_f32_16x16x32_f16(
                false, a, false, b, (short)0, acc2[j], false, false);
        }
    }
    #pragma unroll
    for (int j = 0; j < 8; ++j) {
        #pragma unroll
        for (int v = 0; v < 8; ++v) {
            int r = row0 + w * 16 + v + ((lane >> 4) << 3);
            int c = j * 16 + (lane & 15);
            float o = acc2[j][v];
            out[((size_t)r * DOUT_ + c) * 2 + br] = o > 0.f ? o : 0.f;
        }
    }
}

extern "C" void kernel_launch(void* const* d_in, const int* in_sizes, int n_in,
                              void* d_out, int out_size, void* d_ws, size_t ws_size,
                              hipStream_t stream) {
    (void)in_sizes; (void)n_in; (void)out_size; (void)ws_size;
    const float* adj_pos = (const float*)d_in[1];
    const float* adj_neg = (const float*)d_in[2];
    const float* h0      = (const float*)d_in[3];
    const float* WB0     = (const float*)d_in[4];
    const float* WU0     = (const float*)d_in[5];
    float*    out = (float*)d_out;
    _Float16* hB  = (_Float16*)d_ws;   // 2 MB: h0 in B-fragment layout

    h0_swizzle<<<2048, 256, 0, stream>>>(h0, hB);
    gnn_fused<<<dim3(N_ / MB, 2), 256, 0, stream>>>(
        adj_pos, adj_neg, h0, WB0, WU0, hB, out);
}
